// NonLocalNd_bn_cbam_dc2_31198642438299
// MI455X (gfx1250) — matrix operand order repaired and checked by reading, not compiled, not run on hardware
//
#include <hip/hip_runtime.h>
#include <hip/hip_bf16.h>

// ---------------------------------------------------------------------------
// Problem constants (from the reference): B=2, C=256, H=W=64 -> N=4096, P=128
// ---------------------------------------------------------------------------
#define BB   2
#define CC   256
#define NN   4096
#define PP   128
#define INVSCALE 0.08838834764831845f   // 1/sqrt(128)

typedef __attribute__((ext_vector_type(16))) __bf16 v16bf;
typedef __attribute__((ext_vector_type(8)))  float  v8f;

union FragU { v16bf v; uint4 q[2]; };
union Pack8 { __bf16 h[8]; uint4 q; };

// ---------------------------------------------------------------------------
// WMMA helpers (CDNA5 16x16x32 bf16 -> f32)
// A fragment: 16x32 (MxK), element (m,k) at base[m*ld + k]
//   lane L: m = L&15 ; K = {8h..8h+7} U {8h+16..8h+23}, h = L>=16
// B fragment: 32x16 (KxN), element (k,n) at base[n*ld + k]   (i.e. B^T row-major)
//   lane L: n = L&15 ; K = {8h..8h+7} U {8h+16..8h+23}
// ---------------------------------------------------------------------------
static __device__ __forceinline__ v16bf load_frag_a(const __bf16* base, int ld, int lane) {
  int m  = lane & 15;
  int k0 = (lane & 16) ? 8 : 0;
  const __bf16* p = base + (size_t)m * ld + k0;
  FragU f;
  f.q[0] = *reinterpret_cast<const uint4*>(p);
  f.q[1] = *reinterpret_cast<const uint4*>(p + 16);
  return f.v;
}

static __device__ __forceinline__ v16bf load_frag_b(const __bf16* base, int ld, int lane) {
  int n  = lane & 15;
  int k0 = (lane & 16) ? 8 : 0;
  const __bf16* p = base + (size_t)n * ld + k0;
  FragU f;
  f.q[0] = *reinterpret_cast<const uint4*>(p);
  f.q[1] = *reinterpret_cast<const uint4*>(p + 16);
  return f.v;
}

static __device__ __forceinline__ v8f wmma_bf16(v16bf a, v16bf b, v8f c) {
  // (neg_a, A, neg_b, B, c_mod, C, reuse_a, reuse_b)
  return __builtin_amdgcn_wmma_f32_16x16x32_bf16(false, a, false, b, (short)0, c, false, false);
}

// ---------------------------------------------------------------------------
// 1) 1x1-conv projection:  Y[b,r,n] = W[r,:] . X[b,:,n] (+bias)
//    X:[B,C,N] f32, W:[R,C], Y:[B,R,N] f32.  Coalesced across n; W via s-loads.
// ---------------------------------------------------------------------------
__global__ void proj_kernel(const float* __restrict__ X, const float* __restrict__ W,
                            const float* __restrict__ bias, float* __restrict__ Y, int R) {
  int n = blockIdx.x * blockDim.x + threadIdx.x;
  int r = blockIdx.y, b = blockIdx.z;
  const float* x = X + (size_t)b * CC * NN + n;
  const float* w = W + (size_t)r * CC;
  float acc = bias ? bias[r] : 0.f;
#pragma unroll 8
  for (int c = 0; c < CC; ++c) acc += w[c] * x[(size_t)c * NN];
  Y[((size_t)b * R + r) * NN + n] = acc;
}

// ---------------------------------------------------------------------------
// 2) Optional channel-mean subtraction (over P), bf16 convert, and transpose.
//    In : Yf [B,P,N] f32   Out: OutT [B,N,P] bf16 ; OutPN [B,P,N] bf16 (opt)
// ---------------------------------------------------------------------------
__global__ void whiten_transpose_kernel(const float* __restrict__ Yf,
                                        __bf16* __restrict__ OutT,
                                        __bf16* __restrict__ OutPN, int subMean) {
  int n = blockIdx.x * blockDim.x + threadIdx.x;
  int b = blockIdx.y;
  const float* src = Yf + (size_t)b * PP * NN + n;
  float mean = 0.f;
  if (subMean) {
#pragma unroll 8
    for (int p = 0; p < PP; ++p) mean += src[(size_t)p * NN];
    mean *= (1.f / PP);
  }
  __bf16* dT = OutT + ((size_t)b * NN + n) * PP;
#pragma unroll 8
  for (int p = 0; p < PP; ++p) {
    float v = src[(size_t)p * NN] - mean;
    dT[p] = (__bf16)v;
    if (OutPN) OutPN[((size_t)b * PP + p) * NN + n] = (__bf16)v;
  }
}

__global__ void f2bf_kernel(const float* __restrict__ in, __bf16* __restrict__ out) {
  size_t i = (size_t)blockIdx.x * blockDim.x + threadIdx.x;
  out[i] = (__bf16)in[i];
}

// ---------------------------------------------------------------------------
// 3a) KVT[b,c,p] = sum_m V[b,c,m] * K[b,p,m]   (stage-1 re-association)
//     A = V [16c x 32m] (ld N), B^T = K[P,N] (ld N).  One wave per 16x16 tile.
// ---------------------------------------------------------------------------
__global__ void kvt_kernel(const __bf16* __restrict__ Vb,    // [B,C,N] bf16
                           const __bf16* __restrict__ Kpn,   // [B,P,N] bf16 (whitened)
                           __bf16* __restrict__ KVT) {       // [B,C,P] bf16
  int lane = threadIdx.x & 31;
  int ct = blockIdx.x, pt = blockIdx.y, b = blockIdx.z;
  const __bf16* ab = Vb  + ((size_t)b * CC + ct * 16) * NN;
  const __bf16* bb = Kpn + ((size_t)b * PP + pt * 16) * NN;
  v8f acc = {};
  for (int mk = 0; mk < NN; mk += 32)
    acc = wmma_bf16(load_frag_a(ab + mk, NN, lane), load_frag_b(bb + mk, NN, lane), acc);
  int hi = lane >> 4, p = pt * 16 + (lane & 15);
#pragma unroll
  for (int i = 0; i < 8; ++i)
    KVT[((size_t)b * CC + ct * 16 + i + 8 * hi) * PP + p] = (__bf16)acc[i];
}

// ---------------------------------------------------------------------------
// 3b) out1[b,c,n] = inv * sum_p QT[b,n,p] * KVT[b,c,p]  -> [B,C,N] bf16
// ---------------------------------------------------------------------------
__global__ void qkv1_kernel(const __bf16* __restrict__ QT,   // [B,N,P]
                            const __bf16* __restrict__ KVT,  // [B,C,P]
                            __bf16* __restrict__ Out,        // [B,C,N]
                            float scale) {
  int lane = threadIdx.x & 31;
  int nt = blockIdx.x, ct = blockIdx.y, b = blockIdx.z;
  const __bf16* ab = QT  + ((size_t)b * NN + nt * 16) * PP;
  const __bf16* bb = KVT + ((size_t)b * CC + ct * 16) * PP;
  v8f acc = {};
#pragma unroll
  for (int pk = 0; pk < PP; pk += 32)
    acc = wmma_bf16(load_frag_a(ab + pk, PP, lane), load_frag_b(bb + pk, PP, lane), acc);
  int hi = lane >> 4, ln = lane & 15;
  Pack8 pk8;
#pragma unroll
  for (int i = 0; i < 8; ++i) pk8.h[i] = (__bf16)(acc[i] * scale);
  *reinterpret_cast<uint4*>(Out + ((size_t)b * CC + ct * 16 + ln) * NN + nt * 16 + 8 * hi) = pk8.q;
}

// ---------------------------------------------------------------------------
// 4) Flash attention stage:  O = softmax_m(inv * QT . KT^T) @ V
//    QT,KT: [B,N,P] bf16 ; V,O: [B,C,N] bf16.
//    8 waves/block; wave = (16-row band of 64-row block) x (128-col C half).
//    Online softmax; P tile bounced through per-wave LDS (D-layout -> A-frag).
// ---------------------------------------------------------------------------
__global__ void __launch_bounds__(256)
flash_kernel(const __bf16* __restrict__ QT, const __bf16* __restrict__ KT,
             const __bf16* __restrict__ Vcn, __bf16* __restrict__ Ocn, float scale) {
  __shared__ __align__(16) __bf16 Plds[8][16 * 64];
  int tid = threadIdx.x, lane = tid & 31, w = tid >> 5;
  int rband = w & 3, chalf = w >> 2;
  int b = blockIdx.y;
  int n0 = blockIdx.x * 64 + rband * 16;
  int ln = lane & 15, hi = lane >> 4;

  const __bf16* qb = QT + ((size_t)b * NN + n0) * PP;
  v16bf Aq[4];
#pragma unroll
  for (int ks = 0; ks < 4; ++ks) Aq[ks] = load_frag_a(qb + ks * 32, PP, lane);

  v8f O[8];
#pragma unroll
  for (int j = 0; j < 8; ++j) O[j] = (v8f){};
  float rmax[8], rsum[8];
#pragma unroll
  for (int i = 0; i < 8; ++i) { rmax[i] = -3.0e38f; rsum[i] = 0.f; }

  __bf16* pl = &Plds[w][0];

  for (int mt = 0; mt < NN / 64; ++mt) {
    int m0 = mt * 64;
    if (mt + 1 < NN / 64)   // gfx1250 global_prefetch for next K tile
      __builtin_prefetch(KT + ((size_t)b * NN + m0 + 64) * PP + lane * 8, 0, 3);

    // ----- scores S[16 x 64] = Q . K^T  (16 WMMAs) -----
    v8f S[4];
#pragma unroll
    for (int ct = 0; ct < 4; ++ct) {
      S[ct] = (v8f){};
      const __bf16* kb = KT + ((size_t)b * NN + m0 + ct * 16) * PP;
#pragma unroll
      for (int ks = 0; ks < 4; ++ks)
        S[ct] = wmma_bf16(Aq[ks], load_frag_b(kb + ks * 32, PP, lane), S[ct]);
    }

    // ----- online softmax (row stats live per 16-lane half) -----
#pragma unroll
    for (int i = 0; i < 8; ++i) {
      float t = fmaxf(fmaxf(S[0][i], S[1][i]), fmaxf(S[2][i], S[3][i]));
      t = fmaxf(t, __shfl_xor(t, 1));
      t = fmaxf(t, __shfl_xor(t, 2));
      t = fmaxf(t, __shfl_xor(t, 4));
      t = fmaxf(t, __shfl_xor(t, 8));
      t *= scale;
      float nm   = fmaxf(rmax[i], t);
      float corr = __expf(rmax[i] - nm);
      rmax[i] = nm;
      float ls = 0.f;
#pragma unroll
      for (int ct = 0; ct < 4; ++ct) {
        float pv = __expf(S[ct][i] * scale - nm);
        S[ct][i] = pv;
        ls += pv;
      }
      ls += __shfl_xor(ls, 1);
      ls += __shfl_xor(ls, 2);
      ls += __shfl_xor(ls, 4);
      ls += __shfl_xor(ls, 8);
      rsum[i] = rsum[i] * corr + ls;
#pragma unroll
      for (int j = 0; j < 8; ++j) O[j][i] *= corr;
    }

    // ----- D-layout P -> LDS (bf16), same-wave DS ops stay in order -----
#pragma unroll
    for (int ct = 0; ct < 4; ++ct)
#pragma unroll
      for (int i = 0; i < 8; ++i)
        pl[(i + 8 * hi) * 64 + ct * 16 + ln] = (__bf16)S[ct][i];

    // ----- O += P @ V   (2 k-steps x 8 c-tiles = 16 WMMAs) -----
#pragma unroll
    for (int ks2 = 0; ks2 < 2; ++ks2) {
      v16bf Pa = load_frag_a(pl + ks2 * 32, 64, lane);
#pragma unroll
      for (int j = 0; j < 8; ++j) {
        int c0 = chalf * 128 + j * 16;
        O[j] = wmma_bf16(Pa,
                         load_frag_b(Vcn + ((size_t)b * CC + c0) * NN + m0 + ks2 * 32, NN, lane),
                         O[j]);
      }
    }
  }

  // ----- normalize rows and store as [C,N] bf16 (16B packs) -----
  float rinv[8];
#pragma unroll
  for (int i = 0; i < 8; ++i) rinv[i] = 1.f / rsum[i];
#pragma unroll
  for (int j = 0; j < 8; ++j) {
    Pack8 pk8;
#pragma unroll
    for (int i = 0; i < 8; ++i) pk8.h[i] = (__bf16)(O[j][i] * rinv[i]);
    int c = chalf * 128 + j * 16 + ln;
    *reinterpret_cast<uint4*>(Ocn + ((size_t)b * CC + c) * NN + n0 + 8 * hi) = pk8.q;
  }
}

// ---------------------------------------------------------------------------
// 5) Global-context branch: softmax over N of mask logits, then value.mask
// ---------------------------------------------------------------------------
__global__ void mask_softmax_kernel(const float* __restrict__ logits, float* __restrict__ probs) {
  __shared__ float red[256];
  int b = blockIdx.x, tid = threadIdx.x;
  const float* l = logits + (size_t)b * NN;
  float m = -3.0e38f;
  for (int n = tid; n < NN; n += 256) m = fmaxf(m, l[n]);
  red[tid] = m; __syncthreads();
  for (int s = 128; s > 0; s >>= 1) { if (tid < s) red[tid] = fmaxf(red[tid], red[tid + s]); __syncthreads(); }
  m = red[0]; __syncthreads();
  float sum = 0.f;
  for (int n = tid; n < NN; n += 256) sum += __expf(l[n] - m);
  red[tid] = sum; __syncthreads();
  for (int s = 128; s > 0; s >>= 1) { if (tid < s) red[tid] += red[tid + s]; __syncthreads(); }
  float inv = 1.f / red[0];
  for (int n = tid; n < NN; n += 256) probs[(size_t)b * NN + n] = __expf(l[n] - m) * inv;
}

__global__ void gc_kernel(const float* __restrict__ Vf, const float* __restrict__ probs,
                          float* __restrict__ gcv) {
  __shared__ float red[256];
  int c = blockIdx.x, b = blockIdx.y, tid = threadIdx.x;
  const float* v = Vf + ((size_t)b * CC + c) * NN;
  const float* p = probs + (size_t)b * NN;
  float s = 0.f;
  for (int n = tid; n < NN; n += 256) s += v[n] * p[n];
  red[tid] = s; __syncthreads();
  for (int st = 128; st > 0; st >>= 1) { if (tid < st) red[tid] += red[tid + st]; __syncthreads(); }
  if (tid == 0) gcv[(size_t)b * CC + c] = red[0];
}

// ---------------------------------------------------------------------------
// 6) out = gamma * out3 + out_gc + x     (out3 [B,C,N] == output layout)
// ---------------------------------------------------------------------------
__global__ void final_kernel(const __bf16* __restrict__ out3, const float* __restrict__ gcv,
                             const float* __restrict__ x, const float* __restrict__ gamma,
                             float* __restrict__ out) {
  size_t i = (size_t)blockIdx.x * blockDim.x + threadIdx.x;
  size_t bc = i >> 12;  // / NN
  out[i] = gamma[0] * (float)out3[i] + gcv[bc] + x[i];
}

// ---------------------------------------------------------------------------
// Host-side launcher
// ---------------------------------------------------------------------------
extern "C" void kernel_launch(void* const* d_in, const int* in_sizes, int n_in,
                              void* d_out, int out_size, void* d_ws, size_t ws_size,
                              hipStream_t stream) {
  (void)in_sizes; (void)n_in; (void)out_size; (void)ws_size;
  const float* x     = (const float*)d_in[0];
  const float* x_dc  = (const float*)d_in[1];
  const float* Wq    = (const float*)d_in[2];
  const float* bq    = (const float*)d_in[3];
  const float* Wk    = (const float*)d_in[4];
  const float* bk    = (const float*)d_in[5];
  const float* Wv    = (const float*)d_in[6];
  const float* Wm    = (const float*)d_in[7];
  const float* bm    = (const float*)d_in[8];
  const float* gamma = (const float*)d_in[9];
  float* out = (float*)d_out;

  char* ws = (char*)d_ws;
  size_t off = 0;
  auto take = [&](size_t bytes) -> void* {
    void* p = ws + off;
    off += (bytes + 255) & ~(size_t)255;
    return p;
  };
  float*  stg    = (float*)take((size_t)BB * PP * NN * 4);   // proj staging f32
  float*  valuef = (float*)take((size_t)BB * CC * NN * 4);   // value f32 (gc branch)
  __bf16* qbT    = (__bf16*)take((size_t)BB * NN * PP * 2);  // whitened q  [B,N,P]
  __bf16* kbT    = (__bf16*)take((size_t)BB * NN * PP * 2);  // whitened k  [B,N,P]
  __bf16* qdcbT  = (__bf16*)take((size_t)BB * NN * PP * 2);  // q_dc        [B,N,P]
  __bf16* kdcbT  = (__bf16*)take((size_t)BB * NN * PP * 2);  // k_dc        [B,N,P]
  __bf16* kb_pn  = (__bf16*)take((size_t)BB * PP * NN * 2);  // whitened k  [B,P,N]
  __bf16* vb     = (__bf16*)take((size_t)BB * CC * NN * 2);  // value       [B,C,N]
  __bf16* kvt    = (__bf16*)take((size_t)BB * CC * PP * 2);  // K V^T       [B,C,P]
  __bf16* out1   = (__bf16*)take((size_t)BB * CC * NN * 2);
  __bf16* out2   = (__bf16*)take((size_t)BB * CC * NN * 2);
  __bf16* out3   = (__bf16*)take((size_t)BB * CC * NN * 2);
  float*  mlog   = (float*)take((size_t)BB * NN * 4);
  float*  mprob  = (float*)take((size_t)BB * NN * 4);
  float*  gcv    = (float*)take((size_t)BB * CC * 4);

  dim3 projG(NN / 128, PP, BB), projB(128);
  dim3 whG(NN / 256, BB), whB(256);

  // Projections + whitening / bf16 transposes
  proj_kernel<<<projG, projB, 0, stream>>>(x, Wq, bq, stg, PP);
  whiten_transpose_kernel<<<whG, whB, 0, stream>>>(stg, qbT, nullptr, 1);
  proj_kernel<<<projG, projB, 0, stream>>>(x, Wk, bk, stg, PP);
  whiten_transpose_kernel<<<whG, whB, 0, stream>>>(stg, kbT, kb_pn, 1);
  proj_kernel<<<projG, projB, 0, stream>>>(x_dc, Wq, bq, stg, PP);
  whiten_transpose_kernel<<<whG, whB, 0, stream>>>(stg, qdcbT, nullptr, 0);
  proj_kernel<<<projG, projB, 0, stream>>>(x_dc, Wk, bk, stg, PP);
  whiten_transpose_kernel<<<whG, whB, 0, stream>>>(stg, kdcbT, nullptr, 0);

  proj_kernel<<<dim3(NN / 128, CC, BB), projB, 0, stream>>>(x, Wv, nullptr, valuef, CC);
  f2bf_kernel<<<dim3((BB * CC * NN) / 256), dim3(256), 0, stream>>>(valuef, vb);
  proj_kernel<<<dim3(NN / 128, 1, BB), projB, 0, stream>>>(x, Wm, bm, mlog, 1);

  // Stage 1 (re-associated, no softmax):  out1 = inv * Q^T (K V^T)
  kvt_kernel<<<dim3(CC / 16, PP / 16, BB), dim3(32), 0, stream>>>(vb, kb_pn, kvt);
  qkv1_kernel<<<dim3(NN / 16, CC / 16, BB), dim3(32), 0, stream>>>(qbT, kvt, out1, INVSCALE);

  // Stage 2: out2 = softmax(q_dc^T k) @ out1 ; Stage 3: out3 = softmax(q^T k_dc) @ out2
  flash_kernel<<<dim3(NN / 64, BB), dim3(256), 0, stream>>>(qdcbT, kbT, out1, out2, INVSCALE);
  flash_kernel<<<dim3(NN / 64, BB), dim3(256), 0, stream>>>(qbT, kdcbT, out2, out3, INVSCALE);

  // Global-context branch
  mask_softmax_kernel<<<dim3(BB), dim3(256), 0, stream>>>(mlog, mprob);
  gc_kernel<<<dim3(CC, BB), dim3(256), 0, stream>>>(valuef, mprob, gcv);

  // Final fused add
  final_kernel<<<dim3((BB * CC * NN) / 256), dim3(256), 0, stream>>>(out3, gcv, x, gamma, out);
}
